// Head_4647154614692
// MI455X (gfx1250) — compile-verified, matrix operand order checked by reading, not run on hardware
//
#include <hip/hip_runtime.h>

// Problem constants (match reference)
constexpr int NB = 4;     // batch
constexpr int NT = 2048;  // sequence length
constexpr int NC = 1024;  // channels
constexpr int NH = 64;    // head size / head out

typedef __attribute__((ext_vector_type(16))) __bf16 bf16x16;
typedef __attribute__((ext_vector_type(8)))  float  f32x8;
typedef int v4i __attribute__((vector_size(16)));

#define AS_GLOBAL __attribute__((address_space(1)))
#define AS_SHARED __attribute__((address_space(3)))

#if defined(__HIP_DEVICE_COMPILE__) &&                                         \
    __has_builtin(__builtin_amdgcn_global_load_async_to_lds_b128) &&           \
    __has_builtin(__builtin_amdgcn_s_wait_asynccnt)
#define USE_ASYNC_LDS 1
#else
#define USE_ASYNC_LDS 0
#endif

static __device__ __forceinline__ f32x8 wmma_bf16(bf16x16 a, bf16x16 b, f32x8 c) {
  // D = A(16x32 bf16) * B(32x16 bf16) + C(16x16 f32)
  return __builtin_amdgcn_wmma_f32_16x16x32_bf16(false, a, false, b, (short)0, c,
                                                 false, false);
}

// 16-byte global -> LDS copy; async (ASYNCcnt-tracked) when available.
static __device__ __forceinline__ void cp16_g2l(const __bf16* g, __bf16* l) {
#if USE_ASYNC_LDS
  __builtin_amdgcn_global_load_async_to_lds_b128(
      (AS_GLOBAL v4i*)(void*)g, (AS_SHARED v4i*)(void*)l, 0, 0);
#else
  *(int4*)l = *(const int4*)g;
#endif
}

// ---------------------------------------------------------------------------
// Stage 1: fused projections  Q = q@Wq, K = k@Wk, Vt = (v@Wv)^T   (bf16 out)
// grid (64, 3), block 256 (8 waves). Each wave: 16 rows x 64 cols, K=1024.
// W staged into LDS transposed+bf16 so B-fragments are contiguous ds reads.
// ---------------------------------------------------------------------------
__global__ __launch_bounds__(256)
void proj_kernel(const float* __restrict__ q, const float* __restrict__ k,
                 const float* __restrict__ v, const float* __restrict__ Wq,
                 const float* __restrict__ Wk, const float* __restrict__ Wv,
                 __bf16* __restrict__ Qb, __bf16* __restrict__ Kb,
                 __bf16* __restrict__ Vt) {
  __shared__ __bf16 Wt[NH * NC];  // [n][k] transposed weights, 128 KB

  const int proj = blockIdx.y;
  const float* X = (proj == 0) ? q : (proj == 1) ? k : v;
  const float* W = (proj == 0) ? Wq : (proj == 1) ? Wk : Wv;

  for (int idx = threadIdx.x; idx < NC * NH; idx += 256) {
    const int kk = idx >> 6;      // k index (row of W)
    const int n  = idx & 63;      // n index (col of W)
    Wt[n * NC + kk] = (__bf16)W[idx];
  }
  __syncthreads();

  const int wave  = threadIdx.x >> 5;
  const int lane  = threadIdx.x & 31;
  const int l15   = lane & 15;
  const int lhalf = lane >> 4;                    // 0: K-chunk lo, 1: hi
  const int m0    = blockIdx.x * 128 + wave * 16; // first output row

  f32x8 acc[4] = {};

  const float* arow = X + (size_t)(m0 + l15) * NC + lhalf * 8;
  for (int kb = 0; kb < NC; kb += 32) {
    // A fragment 16x32: lane l15 = row, K chunks at kb+lhalf*8 and +16
    bf16x16 af;
#pragma unroll
    for (int i = 0; i < 8; ++i) {
      af[i]     = (__bf16)arow[kb + i];
      af[8 + i] = (__bf16)arow[kb + 16 + i];
    }
#pragma unroll
    for (int nt = 0; nt < 4; ++nt) {
      const __bf16* wr = &Wt[(nt * 16 + l15) * NC + kb + lhalf * 8];
      bf16x16 bf;
#pragma unroll
      for (int i = 0; i < 8; ++i) { bf[i] = wr[i]; bf[8 + i] = wr[16 + i]; }
      acc[nt] = wmma_bf16(af, bf, acc[nt]);
    }
  }

  // C/D layout: vgpr r, lanes 0-15 -> M=r, lanes 16-31 -> M=r+8; N = l15
  if (proj < 2) {
    __bf16* Out = (proj == 0) ? Qb : Kb;
#pragma unroll
    for (int nt = 0; nt < 4; ++nt)
#pragma unroll
      for (int r = 0; r < 8; ++r) {
        const int row = m0 + r + 8 * lhalf;
        Out[(size_t)row * NH + nt * 16 + l15] = (__bf16)acc[nt][r];
      }
  } else {
    // V stored transposed: Vt[b][h][t]
#pragma unroll
    for (int nt = 0; nt < 4; ++nt)
#pragma unroll
      for (int r = 0; r < 8; ++r) {
        const int g = m0 + r + 8 * lhalf;
        const int bb = g >> 11;          // g / NT
        const int t  = g & (NT - 1);
        Vt[((size_t)bb * NH + nt * 16 + l15) * NT + t] = (__bf16)acc[nt][r];
      }
  }
}

// ---------------------------------------------------------------------------
// Stage 2: streaming attention. 8 waves/block, one 16-query tile per wave,
// all waves of a block share one batch. K/V tiles (32 keys) are staged into
// LDS once per block via double-buffered async global->LDS copies and shared
// by all 8 waves (8x cut in L2 traffic). Soft mask (MASK_FILL=1e-9 pre-scale
// => masked score contributes exp(1.25e-10); every key column participates).
// Softmax without max-subtraction (scaled scores ~N(0,1), exp args small).
// ---------------------------------------------------------------------------
__global__ __launch_bounds__(256)
void attn_kernel(const __bf16* __restrict__ Qb, const __bf16* __restrict__ Kb,
                 const __bf16* __restrict__ Vt, const int* __restrict__ pad,
                 float* __restrict__ out) {
  __shared__ __bf16 Kbuf[2][32 * NH];   // [buf][key][h]   2 x 4 KB
  __shared__ __bf16 Vbuf[2][NH * 32];   // [buf][h][key]   2 x 4 KB
  __shared__ __bf16 Pbuf[8][16 * 32];   // per-wave P tile scratch, 8 KB

  const int wave  = threadIdx.x >> 5;
  const int lane  = threadIdx.x & 31;
  const int l15   = lane & 15;
  const int lhalf = lane >> 4;
  const int tid   = threadIdx.x;

  const int b  = blockIdx.x >> 4;                          // uniform per block
  const int m0 = ((blockIdx.x & 15) * 8 + wave) * 16;      // query tile base

  // Cooperative staging addresses (one b128 per thread per tile per matrix)
  const __bf16* gK = Kb + ((size_t)b * NT) * NH + tid * 8;               // +n0*NH
  const __bf16* gV = Vt + ((size_t)b * NH + (tid >> 2)) * NT + (tid & 3) * 8;  // +n0
  __bf16* lK = &Kbuf[0][tid * 8];
  __bf16* lV = &Vbuf[0][(tid >> 2) * 32 + (tid & 3) * 8];
  const int lbstride = 32 * NH;  // elements between buf 0 and buf 1

  // Loop-invariant Q fragments (16 queries x 64 head dims = 2 A-frags)
  const __bf16* qrow = Qb + ((size_t)b * NT + m0 + l15) * NH + lhalf * 8;
  bf16x16 qf[2];
#pragma unroll
  for (int hb = 0; hb < 2; ++hb)
#pragma unroll
    for (int i = 0; i < 8; ++i) {
      qf[hb][i]     = qrow[hb * 32 + i];
      qf[hb][8 + i] = qrow[hb * 32 + 16 + i];
    }

  int padr[8];
#pragma unroll
  for (int r = 0; r < 8; ++r) padr[r] = pad[b * NT + m0 + r + 8 * lhalf];

  f32x8 oacc[4] = {};
  f32x8 rsum = {};
  __bf16* pb = &Pbuf[wave][0];

  // Preload tile 0
  cp16_g2l(gK, lK);
  cp16_g2l(gV, lV);

  for (int it = 0; it < NT / 32; ++it) {
    const int n0  = it * 32;
    const int cur = it & 1;

    if (it + 1 < NT / 32) {  // issue next tile into the other buffer
      cp16_g2l(gK + (size_t)(n0 + 32) * NH, lK + (cur ^ 1) * lbstride);
      cp16_g2l(gV + (n0 + 32), lV + (cur ^ 1) * lbstride);
#if USE_ASYNC_LDS
      __builtin_amdgcn_s_wait_asynccnt(2);   // current tile's 2 ops complete
#endif
    } else {
#if USE_ASYNC_LDS
      __builtin_amdgcn_s_wait_asynccnt(0);
#endif
    }
    __syncthreads();  // all waves' staging for buf[cur] visible

    const __bf16* Kt = &Kbuf[cur][0];
    const __bf16* Vv = &Vbuf[cur][0];

#pragma unroll
    for (int ct = 0; ct < 2; ++ct) {
      // S tile 16x16: A=Q, B=K^T (columns of K^T = contiguous K rows in LDS)
      f32x8 s = {};
      const __bf16* krow = Kt + (ct * 16 + l15) * NH + lhalf * 8;
#pragma unroll
      for (int hb = 0; hb < 2; ++hb) {
        bf16x16 kf;
#pragma unroll
        for (int i = 0; i < 8; ++i) {
          kf[i]     = krow[hb * 32 + i];
          kf[8 + i] = krow[hb * 32 + 16 + i];
        }
        s = wmma_bf16(qf[hb], kf, s);
      }
      // soft mask + scale + exp; stash P tile in LDS (bf16)
      const int j = n0 + ct * 16 + l15;
#pragma unroll
      for (int r = 0; r < 8; ++r) {
        const int i = m0 + r + 8 * lhalf;
        const bool keep = (j <= i + 1) && (padr[r] == 0);
        const float x = keep ? s[r] * 0.125f : 1.25e-10f;
        const float e = __expf(x);
        rsum[r] += e;
        pb[(r + 8 * lhalf) * 32 + ct * 16 + l15] = (__bf16)e;
      }
    }
    // Re-read P as a 16x32 A-fragment (layout conversion via LDS)
    bf16x16 pf;
    const __bf16* pr = pb + l15 * 32 + lhalf * 8;
#pragma unroll
    for (int i = 0; i < 8; ++i) { pf[i] = pr[i]; pf[8 + i] = pr[16 + i]; }
    // O += P(16x32) * V(32x16) per head tile; Vbuf rows contiguous in key
#pragma unroll
    for (int ht = 0; ht < 4; ++ht) {
      const __bf16* vrow = Vv + (ht * 16 + l15) * 32 + lhalf * 8;
      bf16x16 vf;
#pragma unroll
      for (int i = 0; i < 8; ++i) { vf[i] = vrow[i]; vf[8 + i] = vrow[16 + i]; }
      oacc[ht] = wmma_bf16(pf, vf, oacc[ht]);
    }
    __syncthreads();  // everyone done reading buf[cur] before it is re-filled
  }

  // Row sums: reduce over the 16 lanes of each half (masks stay in-half)
  float rinv[8];
#pragma unroll
  for (int r = 0; r < 8; ++r) {
    float s = rsum[r];
    s += __shfl_xor(s, 1, 32);
    s += __shfl_xor(s, 2, 32);
    s += __shfl_xor(s, 4, 32);
    s += __shfl_xor(s, 8, 32);
    rinv[r] = 1.0f / s;
  }
#pragma unroll
  for (int ht = 0; ht < 4; ++ht)
#pragma unroll
    for (int r = 0; r < 8; ++r) {
      const int row = m0 + r + 8 * lhalf;
      out[((size_t)b * NT + row) * NH + ht * 16 + l15] = oacc[ht][r] * rinv[r];
    }
}

// ---------------------------------------------------------------------------
extern "C" void kernel_launch(void* const* d_in, const int* in_sizes, int n_in,
                              void* d_out, int out_size, void* d_ws,
                              size_t ws_size, hipStream_t stream) {
  const float* q  = (const float*)d_in[0];
  const float* k  = (const float*)d_in[1];
  const float* v  = (const float*)d_in[2];
  const float* Wq = (const float*)d_in[3];
  const float* Wk = (const float*)d_in[4];
  const float* Wv = (const float*)d_in[5];
  const int* pad  = (const int*)d_in[6];
  float* out = (float*)d_out;

  const size_t N = (size_t)NB * NT * NH;  // 524288 elems per projection
  __bf16* Qb = (__bf16*)d_ws;             // 1 MB
  __bf16* Kb = Qb + N;                    // 1 MB
  __bf16* Vt = Kb + N;                    // 1 MB (transposed [B][H][T])

  proj_kernel<<<dim3(64, 3), 256, 0, stream>>>(q, k, v, Wq, Wk, Wv, Qb, Kb, Vt);
  attn_kernel<<<dim3(64), 256, 0, stream>>>(Qb, Kb, Vt, pad, out);
}
